// DCRNNmodel_66133906423961
// MI455X (gfx1250) — matrix operand order repaired and checked by reading, not compiled, not run on hardware
//
#include <hip/hip_runtime.h>

typedef __attribute__((ext_vector_type(16))) __bf16 v16bf;
typedef __attribute__((ext_vector_type(8)))  __bf16 v8bf;
typedef __attribute__((ext_vector_type(8)))  float  v8f;
typedef __attribute__((ext_vector_type(4)))  unsigned int v4u;
typedef __attribute__((ext_vector_type(8)))  int v8i;
typedef __attribute__((ext_vector_type(4)))  int v4i;

namespace {
constexpr int kB   = 64;    // batch
constexpr int kN   = 1024;  // nodes
constexpr int kH   = 64;    // hidden
constexpr int kNM  = 3;     // diffusion matrices
constexpr int kLP  = 128;   // padded inner dim for small GEMM
constexpr int kSeq = 12;
constexpr int kF   = 2;
constexpr int BST  = 72;    // LDS B-tile stride in bf16 (144B rows -> 16B aligned frags)
constexpr int WST  = 136;   // LDS W-tile stride in bf16 (272B rows -> 16B aligned frags)
constexpr int KTS  = 64;    // K-step per barrier pair in big GEMM
constexpr int ATSZ = 64 * KTS;  // A tile: 64 rows x 64 K, bf16 (8KB)
}

// ---------------- WMMA fragment helpers (layouts per CDNA5 ISA 7.12.2) -------

__device__ __forceinline__ v16bf cat8(v8bf lo, v8bf hi) {
  v16bf a;
#pragma unroll
  for (int i = 0; i < 8; ++i) { a[i] = lo[i]; a[i + 8] = hi[i]; }
  return a;
}

// A fragment, 16x32 bf16. rowptr -> &A[row][k0] (row stride given by caller).
// lanes 0-15 hold K 0-7 / 16-23, lanes 16-31 hold K 8-15 / 24-31.
__device__ __forceinline__ v16bf load_a_frag(const __bf16* rowptr, int lanehi) {
  const int o = lanehi << 3;
  v8bf lo = *(const v8bf*)(rowptr + o);
  v8bf hi = *(const v8bf*)(rowptr + 16 + o);
  return cat8(lo, hi);
}

// B fragment, 32x16 bf16, from LDS tile stored [col][k] with row stride `stride`.
// lanes 0-15: K 0-7 / 16-23, lanes 16-31: K 8-15 / 24-31; N = lane&15.
__device__ __forceinline__ v16bf load_b_frag(const __bf16* tile, int col, int lanehi,
                                             int stride) {
  const __bf16* p = tile + col * stride + (lanehi << 3);
  v8bf lo = *(const v8bf*)(p);
  v8bf hi = *(const v8bf*)(p + 16);
  return cat8(lo, hi);
}

__device__ __forceinline__ v8f wmma_bf16(v16bf a, v16bf b, v8f c) {
  return __builtin_amdgcn_wmma_f32_16x16x32_bf16(false, a, false, b, (short)0, c,
                                                 false, false);
}

// ---------------- TDM: async DMA of a 64x64 bf16 tile (row stride 1024) ------
// D# per CDNA5 ISA ch.8: group0 = {flags, lds_addr, global_addr, type=2},
// group1 = {mask/data_size, tensor dims, tile dims, dim0 stride}.
__device__ __forceinline__ void tdm_load_a_tile(const __bf16* gptr,
                                                unsigned lds_byte_off) {
  const unsigned long long ga = (unsigned long long)(uintptr_t)gptr;
  const unsigned TD = 1u << 20;  // generous tensor extents (no OOB clipping)
  v4u g0;
  g0[0] = 1u;                                         // count=1, user descriptor
  g0[1] = lds_byte_off;                               // lds_addr
  g0[2] = (unsigned)(ga & 0xffffffffu);               // global_addr[31:0]
  g0[3] = (unsigned)((ga >> 32) & 0x01ffffffu) | (2u << 30);  // addr[56:32]|type=2
  v8i g1;
  g1[0] = (int)(1u << 16);                            // wg_mask=0, data_size=2B
  g1[1] = (int)((TD & 0xffffu) << 16);                // tensor_dim0[15:0]
  g1[2] = (int)((TD >> 16) | ((TD & 0xffffu) << 16)); // tdim0 hi | tdim1 lo
  g1[3] = (int)((TD >> 16) | ((unsigned)KTS << 16));  // tdim1 hi | tile_dim0=64
  g1[4] = (int)64u;                                   // tile_dim1=64, tile_dim2=0
  g1[5] = (int)1024u;                                 // tensor_dim0_stride = kN
  g1[6] = 0;
  g1[7] = 0;
  v4i z4 = {0, 0, 0, 0};
#if defined(__clang_major__) && (__clang_major__ >= 23)
  v8i z8 = {0, 0, 0, 0, 0, 0, 0, 0};
  __builtin_amdgcn_tensor_load_to_lds(g0, g1, z4, z4, z8, 0);
#else
  __builtin_amdgcn_tensor_load_to_lds(g0, g1, z4, z4, 0);
#endif
}

// ---------------- big GEMM: OUT[b] = act( sum_j S_j @ Z[b,j] + bias ) --------
// MODE 0: NO=128, r|u gates, sigmoid -> RU.   MODE 1: NO=64, c gate,
// tanh + fused GRU update h = u*h + (1-u)*c  (in place on Hbuf).
template <int MODE>
__global__ __launch_bounds__(128) void big_gemm(
    const __bf16* __restrict__ S,      // (kNM, kN, kN) bf16, row-major
    const __bf16* __restrict__ Z,      // (kB, kNM, kN, NO) bf16
    const float*  __restrict__ bias0,  // MODE0: br ; MODE1: bc
    const float*  __restrict__ bias1,  // MODE0: bu ; MODE1: unused
    float*        __restrict__ RU,     // (kB, kN, 128)
    float*        __restrict__ Hbuf)   // MODE1 only: (kB, kN, kH)
{
  constexpr int NO = (MODE == 0) ? 128 : 64;
  __shared__ __bf16 At[2 * ATSZ];    // double-buffered TDM destination (16KB)
  __shared__ __bf16 Bt[64 * BST];    // transposed Z tile, 64 cols x 64 K (9KB)

  const int tid    = threadIdx.x;
  const int lane   = tid & 31;
  const int wave   = tid >> 5;
  const int lanehi = lane >> 4;
  const int nlow   = lane & 15;
  const int bb     = blockIdx.y;
  const int col0   = blockIdx.z * 64;
  const int mbase  = blockIdx.x * 64;        // 64 rows per block (4 waves x 16)

  // Z-tile staging coords: thread covers one k row, 32 consecutive cols.
  const int sk = tid >> 1;
  const int sc = (tid & 1) << 5;

  v8f acc0 = {}, acc1 = {}, acc2 = {}, acc3 = {};

  constexpr int kTotal = kNM * (kN / KTS);  // 48 K-steps of 64
  // S tile for flattened K-step jk: rows [mbase, mbase+64), K [k0, k0+64) of S_j.
  auto s_tile = [&](int jk) {
    const int j = jk >> 4, k0 = (jk & 15) << 6;
    return S + ((size_t)j * kN + (size_t)mbase) * kN + k0;
  };

  if (wave == 0) tdm_load_a_tile(s_tile(0), (unsigned)(uintptr_t)(At));

  for (int jk = 0; jk < kTotal; ++jk) {
    const int j = jk >> 4, k0 = (jk & 15) << 6;
    __syncthreads();  // prior compute done: Bt and the idle At buffer are free
    if (wave == 0) {  // kick next DMA early so it overlaps the Z staging below
      const int nx = (jk + 1 < kTotal) ? jk + 1 : 0;  // dummy keeps wait imm const
      tdm_load_a_tile(s_tile(nx),
                      (unsigned)(uintptr_t)(At + ((jk + 1) & 1) * ATSZ));
    }
    {
      const __bf16* Zj  = Z + (size_t)(bb * kNM + j) * kN * NO;
      const __bf16* src = Zj + (size_t)(k0 + sk) * NO + col0 + sc;
#pragma unroll
      for (int q = 0; q < 4; ++q) {
        v8bf v = *(const v8bf*)(src + q * 8);
#pragma unroll
        for (int e = 0; e < 8; ++e) Bt[(sc + q * 8 + e) * BST + sk] = v[e];
      }
    }
    if (wave == 0)
      __builtin_amdgcn_s_wait_tensorcnt((short)1);  // in-order: tile jk resident
    __syncthreads();  // Bt staged + At[jk&1] DMA-complete, visible block-wide

    const __bf16* Abase = At + (jk & 1) * ATSZ + (size_t)(wave * 16 + nlow) * KTS;
#pragma unroll
    for (int kk = 0; kk < KTS; kk += 32) {
      v16bf a = load_a_frag(Abase + kk, lanehi);
      const __bf16* tk = Bt + kk;
      acc0 = wmma_bf16(a, load_b_frag(tk, nlow,      lanehi, BST), acc0);
      acc1 = wmma_bf16(a, load_b_frag(tk, nlow + 16, lanehi, BST), acc1);
      acc2 = wmma_bf16(a, load_b_frag(tk, nlow + 32, lanehi, BST), acc2);
      acc3 = wmma_bf16(a, load_b_frag(tk, nlow + 48, lanehi, BST), acc3);
    }
  }

#pragma unroll
  for (int r = 0; r < 8; ++r) {
    const int m = mbase + wave * 16 + (lanehi ? 8 : 0) + r;
#pragma unroll
    for (int t = 0; t < 4; ++t) {
      const int c = col0 + t * 16 + nlow;
      const float v =
          (t == 0 ? acc0[r] : t == 1 ? acc1[r] : t == 2 ? acc2[r] : acc3[r]);
      if (MODE == 0) {
        const float bias = (c < kH) ? bias0[c] : bias1[c - kH];
        RU[(size_t)(bb * kN + m) * 128 + c] = 1.0f / (1.0f + __expf(-(v + bias)));
      } else {
        const float cv   = tanhf(v + bias0[c]);
        const size_t hix = (size_t)(bb * kN + m) * kH + c;
        const float u    = RU[(size_t)(bb * kN + m) * 128 + kH + c];
        const float h    = Hbuf[hix];
        Hbuf[hix] = u * h + (1.0f - u) * cv;
      }
    }
  }
}

// ---------------- small GEMM: Z[b,j] = XHpad[b] @ Wpad[j]  (K = 128) ---------
__global__ __launch_bounds__(128) void small_gemm(
    const __bf16* __restrict__ XH,  // (kB, kN, kLP)
    const __bf16* __restrict__ W,   // (kNM, kLP, NO)
    __bf16*       __restrict__ Z,   // (kB, kNM, kN, NO)
    int NO, int nstrips)
{
  __shared__ __bf16 Wl[64 * WST];
  const int tid    = threadIdx.x;
  const int lane   = tid & 31;
  const int wave   = tid >> 5;
  const int lanehi = lane >> 4;
  const int nlow   = lane & 15;
  const int mt     = blockIdx.x * 4 + wave;
  const int bb     = blockIdx.y;
  const int j      = blockIdx.z / nstrips;
  const int strip  = blockIdx.z % nstrips;
  const int col0   = strip * 64;

  // stage W[j][k=tid][col0..col0+64) into Wl[col][k]
  {
    const __bf16* src = W + ((size_t)j * kLP + tid) * NO + col0;
#pragma unroll
    for (int c8 = 0; c8 < 8; ++c8) {
      v8bf q = *(const v8bf*)(src + c8 * 8);
#pragma unroll
      for (int e = 0; e < 8; ++e) Wl[(c8 * 8 + e) * WST + tid] = q[e];
    }
  }
  __syncthreads();

  const int arow = mt * 16 + nlow;
  const __bf16* Arow = XH + ((size_t)bb * kN + arow) * kLP;
  v8f acc0 = {}, acc1 = {}, acc2 = {}, acc3 = {};
#pragma unroll
  for (int k0 = 0; k0 < kLP; k0 += 32) {
    v16bf a = load_a_frag(Arow + k0, lanehi);
    const __bf16* tk = Wl + k0;
    acc0 = wmma_bf16(a, load_b_frag(tk, nlow,      lanehi, WST), acc0);
    acc1 = wmma_bf16(a, load_b_frag(tk, nlow + 16, lanehi, WST), acc1);
    acc2 = wmma_bf16(a, load_b_frag(tk, nlow + 32, lanehi, WST), acc2);
    acc3 = wmma_bf16(a, load_b_frag(tk, nlow + 48, lanehi, WST), acc3);
  }

#pragma unroll
  for (int r = 0; r < 8; ++r) {
    const int m = mt * 16 + (lanehi ? 8 : 0) + r;
    __bf16* zr = Z + ((size_t)(bb * kNM + j) * kN + m) * NO + col0;
#pragma unroll
    for (int t = 0; t < 4; ++t) {
      const float v =
          (t == 0 ? acc0[r] : t == 1 ? acc1[r] : t == 2 ? acc2[r] : acc3[r]);
      zr[t * 16 + nlow] = (__bf16)v;
    }
  }
}

// ---------------- data marshalling -------------------------------------------

// XHpad[b,m,l] = l<Lx ? x : (l<Lx+H ? h*(r?) : 0), bf16.
__global__ void pack_xh(const float* __restrict__ xsrc, long long xbs, int xrs,
                        int Lx, const float* __restrict__ h,
                        const float* __restrict__ r, __bf16* __restrict__ XH) {
  const size_t i = (size_t)blockIdx.x * blockDim.x + threadIdx.x;
  const int l = (int)(i & (kLP - 1));
  const size_t bm = i >> 7;
  const int m = (int)(bm & (kN - 1));
  const int b = (int)(bm >> 10);
  float v = 0.0f;
  if (l < Lx) {
    v = xsrc[(size_t)b * xbs + (size_t)m * xrs + l];
  } else if (l < Lx + kH) {
    const int o = l - Lx;
    v = h[(size_t)(b * kN + m) * kH + o];
    if (r) v *= r[(size_t)(b * kN + m) * 128 + o];
  }
  XH[i] = (__bf16)v;
}

// Wpad[j,l,o] bf16 from [Wa | Wb] (each (kNM, L, kH) f32), zero-padded to kLP.
__global__ void pack_w(const float* __restrict__ Wa, const float* __restrict__ Wb,
                       int L, __bf16* __restrict__ dst, int NO) {
  const int i = blockIdx.x * blockDim.x + threadIdx.x;
  if (i >= kNM * kLP * NO) return;
  const int o = i % NO;
  const int l = (i / NO) % kLP;
  const int j = i / (NO * kLP);
  float v = 0.0f;
  if (l < L) {
    v = (o < kH) ? Wa[((size_t)j * L + l) * kH + o]
                 : Wb[((size_t)j * L + l) * kH + (o - kH)];
  }
  dst[i] = (__bf16)v;
}

__global__ void cvt_bf16(const float* __restrict__ s, __bf16* __restrict__ d,
                         size_t n) {
  const size_t i = (size_t)blockIdx.x * blockDim.x + threadIdx.x;
  if (i < n) d[i] = (__bf16)s[i];
}

// out[b,t,n] = h[b,n,:] . proj_w + proj_b ; also feeds decoder input.
__global__ void proj_kernel(const float* __restrict__ h,
                            const float* __restrict__ pw,
                            const float* __restrict__ pb,
                            float* __restrict__ out, float* __restrict__ din,
                            int t) {
  const int i = blockIdx.x * blockDim.x + threadIdx.x;  // b*kN + n
  const float* hr = h + (size_t)i * kH;
  float s = pb[0];
#pragma unroll
  for (int o = 0; o < kH; ++o) s += hr[o] * pw[o];
  const int b = i >> 10, n = i & (kN - 1);
  out[(size_t)(b * kSeq + t) * kN + n] = s;
  din[i] = s;
}

// ---------------- host orchestration -----------------------------------------

extern "C" void kernel_launch(void* const* d_in, const int* in_sizes, int n_in,
                              void* d_out, int out_size, void* d_ws,
                              size_t ws_size, hipStream_t stream) {
  (void)in_sizes; (void)n_in; (void)out_size; (void)ws_size;
  const float* x = (const float*)d_in[0];
  const float* sup = (const float*)d_in[1];
  const float *Wr[4], *Wu[4], *Wc[4], *br_[4], *bu_[4], *bc_[4];
  int idx = 2;
  for (int l = 0; l < 4; ++l) {   // enc0, enc1, dec0, dec1
    Wr[l]  = (const float*)d_in[idx++];
    br_[l] = (const float*)d_in[idx++];
    Wu[l]  = (const float*)d_in[idx++];
    bu_[l] = (const float*)d_in[idx++];
    Wc[l]  = (const float*)d_in[idx++];
    bc_[l] = (const float*)d_in[idx++];
  }
  const float* pw = (const float*)d_in[idx++];
  const float* pb = (const float*)d_in[idx++];
  const int Ldin[4] = {kF + kH, kH + kH, 1 + kH, kH + kH};

  char* p = (char*)d_ws;
  auto take = [&](size_t bytes) {
    char* q = p;
    p += (bytes + 255) & ~(size_t)255;
    return q;
  };
  __bf16* Sbf = (__bf16*)take((size_t)kNM * kN * kN * 2);
  __bf16 *Wru[4], *Wcp[4];
  for (int l = 0; l < 4; ++l) Wru[l] = (__bf16*)take((size_t)kNM * kLP * 128 * 2);
  for (int l = 0; l < 4; ++l) Wcp[l] = (__bf16*)take((size_t)kNM * kLP * 64 * 2);
  __bf16* XH = (__bf16*)take((size_t)kB * kN * kLP * 2);
  __bf16* Zb = (__bf16*)take((size_t)kB * kNM * kN * 128 * 2);
  float* RU  = (float*)take((size_t)kB * kN * 128 * 4);
  float* h0  = (float*)take((size_t)kB * kN * kH * 4);
  float* h1  = (float*)take((size_t)kB * kN * kH * 4);
  float* din = (float*)take((size_t)kB * kN * 4);

  // per-launch prep (deterministic, capture-safe)
  {
    const size_t ns = (size_t)kNM * kN * kN;
    cvt_bf16<<<dim3((unsigned)((ns + 255) / 256)), 256, 0, stream>>>(sup, Sbf, ns);
    for (int l = 0; l < 4; ++l) {
      pack_w<<<dim3((kNM * kLP * 128 + 255) / 256), 256, 0, stream>>>(
          Wr[l], Wu[l], Ldin[l], Wru[l], 128);
      pack_w<<<dim3((kNM * kLP * 64 + 255) / 256), 256, 0, stream>>>(
          Wc[l], nullptr, Ldin[l], Wcp[l], 64);
    }
    hipMemsetAsync(h0, 0, (size_t)kB * kN * kH * 4, stream);
    hipMemsetAsync(h1, 0, (size_t)kB * kN * kH * 4, stream);
    hipMemsetAsync(din, 0, (size_t)kB * kN * 4, stream);
  }

  const unsigned npack = (unsigned)(((size_t)kB * kN * kLP) / 256);
  auto gru = [&](const float* xsrc, long long xbs, int xrs, int Lx, float* hbuf,
                 int l) {
    pack_xh<<<npack, 256, 0, stream>>>(xsrc, xbs, xrs, Lx, hbuf, nullptr, XH);
    small_gemm<<<dim3(16, kB, kNM * 2), 128, 0, stream>>>(XH, Wru[l], Zb, 128, 2);
    big_gemm<0><<<dim3(16, kB, 2), 128, 0, stream>>>(Sbf, Zb, br_[l], bu_[l], RU,
                                                     nullptr);
    pack_xh<<<npack, 256, 0, stream>>>(xsrc, xbs, xrs, Lx, hbuf, RU, XH);
    small_gemm<<<dim3(16, kB, kNM), 128, 0, stream>>>(XH, Wcp[l], Zb, 64, 1);
    big_gemm<1><<<dim3(16, kB, 1), 128, 0, stream>>>(Sbf, Zb, bc_[l], nullptr, RU,
                                                     hbuf);
  };

  // encoder (layer-lockstep per timestep)
  for (int t = 0; t < kSeq; ++t) {
    gru(x + (size_t)t * kN * kF, (long long)(2 * kSeq) * kN * kF, kF, kF, h0, 0);
    gru(h0, (long long)kN * kH, kH, kH, h1, 1);
  }
  // decoder: h0/h1 carry over as initial states; first input is zeros.
  for (int t = 0; t < kSeq; ++t) {
    gru(din, (long long)kN, 1, 1, h0, 2);
    gru(h0, (long long)kN * kH, kH, kH, h1, 3);
    proj_kernel<<<(kB * kN) / 256, 256, 0, stream>>>(h1, pw, pb, (float*)d_out,
                                                     din, t);
  }
}
